// TextureMixDecoder_17205638987821
// MI455X (gfx1250) — matrix operand order repaired and checked by reading, not compiled, not run on hardware
//
#include <hip/hip_runtime.h>

// CDNA5 / gfx1250 — wave32, WMMA 16x16x32 f16->f32.
typedef __attribute__((ext_vector_type(16))) _Float16 v16h;
typedef __attribute__((ext_vector_type(8)))  float    v8f;
typedef __attribute__((ext_vector_type(2)))  float    f2v;
typedef __attribute__((ext_vector_type(4)))  float    f4v;

#define TEX_T 32
#define TEX_C 7
#define TEX_R 512

// Possibly-4B-aligned float2 load (x-corner pair); memcpy lets the backend
// emit a single global_load_b64 (unaligned-access-mode) instead of 2x b32.
__device__ __forceinline__ f2v ldg2(const float* p) {
  f2v r;
  __builtin_memcpy(&r, p, sizeof(r));
  return r;
}

// Non-temporal b128 load for the single-use mix-weight stream: keep the
// 192MB L2 reserved for the 234MB texture working set.
__device__ __forceinline__ f4v ldg4_nt(const float* p) {
  return __builtin_nontemporal_load((const f4v*)p);
}

__global__ __launch_bounds__(256) void texmix_wmma(
    const float* __restrict__ texcoord,  // (N,2)
    const float* __restrict__ mixw,      // (N,32)
    const float* __restrict__ tex,       // (32,7,512,512)
    float* __restrict__ out,             // (N,7)
    int N)
{
  const int lane = (int)(threadIdx.x & 31u);
  const int wave = (int)((blockIdx.x * blockDim.x + threadIdx.x) >> 5);
  const int tile = wave;                       // 16 samples per wave
  if (tile * 16 >= N) return;                  // uniform across the wave

  const int m    = lane & 15;                  // this lane's sample-in-tile
  const int half = lane >> 4;                  // 0: t=0..15, 1: t=16..31
  const int s    = tile * 16 + m;
  const int sl   = (s < N) ? s : (N - 1);      // clamp for loads only

  // ---------------- per-sample bilinear setup (dup'd in both lane halves) ---
  f2v tc = ldg2(texcoord + (size_t)sl * 2);
  float ix = ((tc.x + 1.0f) * (float)TEX_R - 1.0f) * 0.5f;
  float iy = ((tc.y + 1.0f) * (float)TEX_R - 1.0f) * 0.5f;
  float x0f = floorf(ix), y0f = floorf(iy);
  int   x0  = (int)x0f,   y0  = (int)y0f;      // x0 in [-1,511], y0 likewise
  float fx  = ix - x0f,   fy  = iy - y0f;
  float omfx = 1.0f - fx, omfy = 1.0f - fy;

  bool vx0 = (x0 >= 0),  vx1 = (x0 <= TEX_R - 2);
  bool vy0 = (y0 >= 0),  vy1 = (y0 <= TEX_R - 2);
  float w00 = (vx0 && vy0) ? omfx * omfy : 0.0f;
  float w10 = (vx1 && vy0) ? fx   * omfy : 0.0f;
  float w01 = (vx0 && vy1) ? omfx * fy   : 0.0f;
  float w11 = (vx1 && vy1) ? fx   * fy   : 0.0f;

  // Branchless border handling: one b64 load at clamped xb covers both
  // x-corners; edge cases fixed up by per-lane selects (invalid corner has
  // weight 0, so its value is don't-care).
  int xb   = min(max(x0, 0), TEX_R - 2);
  int rowA = min(max(y0,     0), TEX_R - 1);
  int rowB = min(max(y0 + 1, 0), TEX_R - 1);
  bool xsel_lo = (x0 >= TEX_R - 1);            // x0 corner lives in .y
  bool xsel_hi = (x0 < 0);                     // x1 corner lives in .x
  const size_t offA = (size_t)rowA * TEX_R + (size_t)xb;
  const size_t offB = (size_t)rowB * TEX_R + (size_t)xb;
  const int tbase = half * 16;                 // B-operand K range for this lane

  // ---------------- A fragment: 16x32 mix-weight block (f16) ---------------
  // A layout (contiguous, matches B): lane<16 elems 0-15 -> K 0-15;
  // lane>=16: K 16-31.  All loads NT (streamed once).
  const float* mwrow = mixw + (size_t)sl * TEX_T;
  const int kb0 = half ? 16 : 0;
  f4v q0 = ldg4_nt(mwrow + kb0);
  f4v q1 = ldg4_nt(mwrow + kb0 + 4);
  f4v q2 = ldg4_nt(mwrow + kb0 + 8);
  f4v q3 = ldg4_nt(mwrow + kb0 + 12);
  v16h a;
  a[0]  = (_Float16)q0.x; a[1]  = (_Float16)q0.y; a[2]  = (_Float16)q0.z; a[3]  = (_Float16)q0.w;
  a[4]  = (_Float16)q1.x; a[5]  = (_Float16)q1.y; a[6]  = (_Float16)q1.z; a[7]  = (_Float16)q1.w;
  a[8]  = (_Float16)q2.x; a[9]  = (_Float16)q2.y; a[10] = (_Float16)q2.z; a[11] = (_Float16)q2.w;
  a[12] = (_Float16)q3.x; a[13] = (_Float16)q3.y; a[14] = (_Float16)q3.z; a[15] = (_Float16)q3.w;

  // ---------------- per channel: gather 16 t-values, 1 WMMA, diag store -----
  for (int c = 0; c < TEX_C; ++c) {
    v16h b;
#pragma unroll
    for (int tt = 0; tt < 16; ++tt) {
      const float* p = tex +
          (size_t)((tbase + tt) * TEX_C + c) * (size_t)(TEX_R * TEX_R);
      f2v ra = ldg2(p + offA);                 // row y0: (x0,x1) pair
      f2v rb = ldg2(p + offB);                 // row y1: (x0,x1) pair
      float v0a = xsel_lo ? ra.y : ra.x;
      float v1a = xsel_hi ? ra.x : ra.y;
      float v0b = xsel_lo ? rb.y : rb.x;
      float v1b = xsel_hi ? rb.x : rb.y;
      float v = w00 * v0a + w10 * v1a + w01 * v0b + w11 * v1b;
      b[tt] = (_Float16)v;                     // B layout: lane<16 K=tt, lane>=16 K=16+tt
    }

    v8f d = {};
    // D(16x16) = MW(16x32) x V^T(32x16); we need only the diagonal.
    d = __builtin_amdgcn_wmma_f32_16x16x32_f16(
        /*neg_a=*/false, a, /*neg_b=*/false, b,
        /*c_mod=*/(short)0, d, /*reuse_a=*/false, /*reuse_b=*/false);

    // Diagonal D[n',n']: n'<8 -> lane n', vgpr n'; n'>=8 -> lane n'+16, vgpr n'-8.
    int r = (lane < 16) ? lane : (lane - 24);
    float val = d[0];
    val = (r == 1) ? d[1] : val;
    val = (r == 2) ? d[2] : val;
    val = (r == 3) ? d[3] : val;
    val = (r == 4) ? d[4] : val;
    val = (r == 5) ? d[5] : val;
    val = (r == 6) ? d[6] : val;
    val = (r == 7) ? d[7] : val;
    if (((lane < 8) || (lane >= 24)) && (s < N)) {
      __builtin_nontemporal_store(val, out + (size_t)s * TEX_C + c);
    }
  }
}

extern "C" void kernel_launch(void* const* d_in, const int* in_sizes, int n_in,
                              void* d_out, int out_size, void* d_ws, size_t ws_size,
                              hipStream_t stream) {
  const float* texcoord = (const float*)d_in[0];  // (N,2)   f32
  const float* mixw     = (const float*)d_in[1];  // (N,32)  f32
  const float* tex      = (const float*)d_in[2];  // (32,7,512,512) f32
  float* out            = (float*)d_out;          // (N,7)   f32

  const int N      = in_sizes[0] / 2;
  const int tiles  = (N + 15) / 16;               // 16 samples per wave
  const int wavesB = 8;                           // 256 threads = 8 wave32
  const int blocks = (tiles + wavesB - 1) / wavesB;
  texmix_wmma<<<blocks, 256, 0, stream>>>(texcoord, mixw, tex, out, N);
}
